// FluxUNet_64003602645716
// MI455X (gfx1250) — matrix operand order repaired and checked by reading, not compiled, not run on hardware
//
#include <hip/hip_runtime.h>

typedef _Float16 h16;
typedef __attribute__((ext_vector_type(16))) _Float16 v16h;
typedef __attribute__((ext_vector_type(8)))  float    v8f;

// ---------------- problem constants ----------------
#define BB   8
#define CC   512
#define HWn  1024
#define SS   77
#define EE   768
#define NH   8
#define HD   64
#define HID  2048
#define LV   1101      // valid sequence length (1024 spatial + 77 cond)
#define LP   1120      // padded to multiple of 32 (35 key tiles)

// ---------------- WMMA fragment index maps (this code's contiguous k order, 16-bit; A and B match) ----------------
__device__ __forceinline__ int a_k0(int j, int hi) {
  // A 16x32, this code's order (not ISA table): VGPR j holds K=2j,2j+1 (lanes0-15), K=16+2j,+1 (lanes16-31)
  return 2 * j + 16 * hi;
}
__device__ __forceinline__ int b_k0(int j, int hi) {
  // B 32x16, same order as A (not ISA table): VGPR j holds K=2j,2j+1 (lanes0-15), K=16+2j,+1 (lanes16-31)
  return 2 * j + 16 * hi;
}

// ---------------- CDNA5 async global->LDS copy (ASYNCcnt) ----------------
// LDS offset = low 32 bits of the flat shared-memory address (ISA 10.2).
__device__ __forceinline__ void async_ld_b128(const void* g, const void* l) {
  unsigned loff = (unsigned)(unsigned long long)l;
  unsigned long long ga = (unsigned long long)g;
  asm volatile("global_load_async_to_lds_b128 %0, %1, off"
               :: "v"(loff), "v"(ga) : "memory");
}
__device__ __forceinline__ void wait_async0() {
  asm volatile("s_wait_asynccnt 0" ::: "memory");
}

// ---------------- small elementwise kernels ----------------
__global__ void k_cvt_f32_f16(const float* __restrict__ in, h16* __restrict__ out, int n) {
  int i = blockIdx.x * 256 + threadIdx.x;
  if (i < n) out[i] = (h16)in[i];
}

__global__ void k_silu(const float* __restrict__ in, float* __restrict__ out, int n) {
  int i = blockIdx.x * 256 + threadIdx.x;
  if (i < n) { float v = in[i]; out[i] = v / (1.f + __expf(-v)); }
}

// mod = silu(femb) @ mod_w + mod_b   -> [B, 6C]
__global__ void k_mod(const float* __restrict__ sf, const float* __restrict__ W,
                      const float* __restrict__ bias, float* __restrict__ out) {
  int idx = blockIdx.x * 256 + threadIdx.x;          // B*3072
  if (idx >= BB * 6 * CC) return;
  int b = idx / (6 * CC), o = idx % (6 * CC);
  const float* s = sf + b * EE;
  float acc = bias[o];
  for (int e = 0; e < EE; ++e) acc += s[e] * W[e * (6 * CC) + o];
  out[idx] = acc;
}

// channel RMSNorm + adaLN modulate, writes f16 [B][C][HW]
__global__ void k_normmod(const float* __restrict__ x, const float* __restrict__ nw,
                          const float* __restrict__ modv, int shOff, int scOff,
                          h16* __restrict__ out) {
  int idx = blockIdx.x * 256 + threadIdx.x;          // B*HW
  if (idx >= BB * HWn) return;
  int b = idx >> 10, n = idx & (HWn - 1);
  const float* xb = x + ((long long)b * CC * HWn);
  float ss = 0.f;
  for (int c = 0; c < CC; ++c) { float v = xb[c * HWn + n]; ss += v * v; }
  float r = rsqrtf(ss * (1.f / CC) + 1e-6f);
  const float* mb = modv + b * (6 * CC);
  h16* ob = out + ((long long)b * CC * HWn);
  for (int c = 0; c < CC; ++c) {
    float v = xb[c * HWn + n] * r * nw[c];
    v = v * (1.f + mb[scOff + c]) + mb[shOff + c];
    ob[c * HWn + n] = (h16)v;
  }
}

// out[b][c][n] = xin[b][c][n] + mod[b][gOff+c] * Y[b][n][c]
__global__ void k_resid(const float* __restrict__ xin, const float* __restrict__ modv,
                        int gOff, const float* __restrict__ Y, float* __restrict__ out) {
  long long idx = (long long)blockIdx.x * 256 + threadIdx.x;
  if (idx >= (long long)BB * CC * HWn) return;
  int n = (int)(idx & (HWn - 1));
  int c = (int)((idx >> 10) & (CC - 1));
  int b = (int)(idx >> 19);
  float g = modv[b * (6 * CC) + gOff + c];
  float y = Y[((long long)b * HWn + n) * CC + c];
  out[idx] = xin[idx] + g * y;
}

// ---------------- RoPE + QK RMSNorm, builds padded f16 Q/K/V [B][NH][LP][HD] ----------------
// qkv_s is [B][HW][3C] row-major, qkv_c is [B][S][3C] row-major (identical indexing).
__global__ __launch_bounds__(256) void k_build_qkv(
    const float* __restrict__ qs, const float* __restrict__ qc,
    const float* __restrict__ freqs,   // [HD/2][2]
    const float* __restrict__ qn, const float* __restrict__ kn,
    h16* __restrict__ Qo, h16* __restrict__ Ko, h16* __restrict__ Vo) {
  int gw = (blockIdx.x * 256 + threadIdx.x) >> 5;  // one wave per (b,h,row)
  int lane = threadIdx.x & 31;
  int row = gw % LP, bh = gw / LP;
  if (bh >= BB * NH) return;
  int h = bh & (NH - 1), b = bh >> 3;
  long long ob = ((long long)bh * LP + row) * HD + 2 * lane;
  if (row >= LV) {                                  // zero the padding rows
    h16 z = (h16)0.f;
    Qo[ob] = z; Qo[ob + 1] = z; Ko[ob] = z; Ko[ob + 1] = z; Vo[ob] = z; Vo[ob + 1] = z;
    return;
  }
  int d = 2 * lane;
  float cv, sv;
  const float* base;
  if (row < HWn) {                                  // spatial token: 2D RoPE
    base = qs + ((long long)b * HWn + row) * (3 * CC);
    float gy = (float)(row >> 5), gx = (float)(row & 31);
    float ang = gy * freqs[2 * lane] + gx * freqs[2 * lane + 1];
    cv = cosf(ang); sv = sinf(ang);
  } else {                                          // cond token: 1D RoPE
    int s = row - HWn;
    base = qc + ((long long)b * SS + s) * (3 * CC);
    float inv = powf(10000.f, -(float)d * (1.f / HD));
    float ang = (float)s * inv;
    cv = cosf(ang); sv = sinf(ang);
  }
  float q0 = base[h * HD + d],          q1 = base[h * HD + d + 1];
  float k0 = base[CC + h * HD + d],     k1 = base[CC + h * HD + d + 1];
  float v0 = base[2 * CC + h * HD + d], v1 = base[2 * CC + h * HD + d + 1];
  float rq0 = q0 * cv - q1 * sv, rq1 = q1 * cv + q0 * sv;
  float rk0 = k0 * cv - k1 * sv, rk1 = k1 * cv + k0 * sv;
  float sq = rq0 * rq0 + rq1 * rq1, sk = rk0 * rk0 + rk1 * rk1;
#pragma unroll
  for (int m = 1; m < 32; m <<= 1) { sq += __shfl_xor(sq, m, 32); sk += __shfl_xor(sk, m, 32); }
  const float eps = 1.1920929e-07f;
  float riq = rsqrtf(sq * (1.f / HD) + eps), rik = rsqrtf(sk * (1.f / HD) + eps);
  Qo[ob] = (h16)(rq0 * riq * qn[d]); Qo[ob + 1] = (h16)(rq1 * riq * qn[d + 1]);
  Ko[ob] = (h16)(rk0 * rik * kn[d]); Ko[ob + 1] = (h16)(rk1 * rik * kn[d + 1]);
  Vo[ob] = (h16)v0;                  Vo[ob + 1] = (h16)v1;
}

// ---------------- generic WMMA GEMM with async double-buffered LDS staging ----------------
// A_KM: 0 -> A is [M,K] row-major (LDS tile [m][k]); 1 -> A is [K,M] (LDS tile [k][m])
// B_NK: 0 -> B is [K,N] row-major (LDS tile [k][n]); 1 -> B is [N,K] (LDS tile [n][k])
// EPI : 0 none, 1 bias, 2 bias + tanh-GELU
template <int A_KM, int B_NK, int EPI>
__global__ __launch_bounds__(256) void gemm_wmma(
    const h16* __restrict__ A, long long sA,
    const h16* __restrict__ Bm, long long sB,
    float* __restrict__ Cf, long long sCf,
    h16* __restrict__ Ch, long long sCh,
    const float* __restrict__ bias, int M, int N, int K) {
  __shared__ h16 As[2][2048];
  __shared__ h16 Bs[2][2048];
  const int tid = threadIdx.x;
  const int lane = tid & 31, wave = tid >> 5;
  const int wm = wave & 3, wn = wave >> 2;     // wave tile: rows wm*16, cols wn*32
  const int ml = lane & 15, hi = lane >> 4;
  const int n0 = blockIdx.x * 64, m0 = blockIdx.y * 64;
  const int batch = blockIdx.z;
  A += (long long)batch * sA;
  Bm += (long long)batch * sB;

  // per-thread staging coordinates (each thread copies one 16B chunk per matrix)
  int a_r, a_c, b_r, b_c;
  if (A_KM == 0) { int idx = tid * 8; a_r = idx >> 5; a_c = idx & 31; }   // [m][k]
  else           { a_r = tid & 31;    a_c = (tid >> 5) * 8; }             // [k][m]
  if (B_NK == 0) { int idx = tid * 8; b_r = idx >> 6; b_c = idx & 63; }   // [k][n]
  else           { b_r = tid & 63;    b_c = (tid >> 6) * 8; }             // [n][k]

  auto stage = [&](int kt, int buf) {
    if (A_KM == 0) {
      int grow = m0 + a_r; if (grow >= M) grow = M - 1;  // garbage rows never stored
      async_ld_b128(A + (long long)grow * K + kt + a_c, &As[buf][a_r * 32 + a_c]);
    } else {
      int gm = m0 + a_c; if (gm + 8 > M) gm = M - 8;
      async_ld_b128(A + (long long)(kt + a_r) * M + gm, &As[buf][a_r * 64 + a_c]);
    }
    if (B_NK == 0) {
      async_ld_b128(Bm + (long long)(kt + b_r) * N + n0 + b_c, &Bs[buf][b_r * 64 + b_c]);
    } else {
      int gn = n0 + b_r; if (gn >= N) gn = N - 1;        // garbage cols never stored
      async_ld_b128(Bm + (long long)gn * K + kt + b_c, &Bs[buf][b_r * 32 + b_c]);
    }
  };

  v8f acc0, acc1;
#pragma unroll
  for (int r = 0; r < 8; ++r) { acc0[r] = 0.f; acc1[r] = 0.f; }

  const int nk = K >> 5;
  stage(0, 0);
  wait_async0();
  __syncthreads();
  for (int t = 0; t < nk; ++t) {
    const int cur = t & 1;
    if (t + 1 < nk) stage((t + 1) << 5, cur ^ 1);   // prefetch next tile (async)
    const h16* Ab = As[cur];
    const h16* Bb = Bs[cur];
    v16h a, b0, b1;
#pragma unroll
    for (int j = 0; j < 8; ++j) {
      int ka = a_k0(j, hi);
      if (A_KM == 0) {
        int off = (wm * 16 + ml) * 32 + ka;
        a[2 * j] = Ab[off]; a[2 * j + 1] = Ab[off + 1];
      } else {
        int mm = wm * 16 + ml;
        a[2 * j] = Ab[ka * 64 + mm]; a[2 * j + 1] = Ab[(ka + 1) * 64 + mm];
      }
      int kb = b_k0(j, hi);
      int c0 = wn * 32 + ml, c1 = c0 + 16;
      if (B_NK == 0) {
        b0[2 * j] = Bb[kb * 64 + c0]; b0[2 * j + 1] = Bb[(kb + 1) * 64 + c0];
        b1[2 * j] = Bb[kb * 64 + c1]; b1[2 * j + 1] = Bb[(kb + 1) * 64 + c1];
      } else {
        b0[2 * j] = Bb[c0 * 32 + kb]; b0[2 * j + 1] = Bb[c0 * 32 + kb + 1];
        b1[2 * j] = Bb[c1 * 32 + kb]; b1[2 * j + 1] = Bb[c1 * 32 + kb + 1];
      }
    }
    acc0 = __builtin_amdgcn_wmma_f32_16x16x32_f16(false, a, false, b0, (short)0, acc0, false, false);
    acc1 = __builtin_amdgcn_wmma_f32_16x16x32_f16(false, a, false, b1, (short)0, acc1, false, false);
    wait_async0();
    __syncthreads();
  }
  // ---- epilogue ----
#pragma unroll
  for (int t2 = 0; t2 < 2; ++t2) {
    v8f acc = t2 ? acc1 : acc0;
#pragma unroll
    for (int r = 0; r < 8; ++r) {
      int row = m0 + wm * 16 + r + 8 * hi;
      int col = n0 + wn * 32 + t2 * 16 + ml;
      if (row < M) {
        float v = acc[r];
        if (EPI >= 1) v += bias[col];
        if (EPI == 2) {
          float u = v, t = 0.7978845608f * (u + 0.044715f * u * u * u);
          v = 0.5f * u * (1.f + tanhf(t));
        }
        if (Cf) Cf[(long long)batch * sCf + (long long)row * N + col] = v;
        if (Ch) Ch[(long long)batch * sCh + (long long)row * N + col] = (h16)v;
      }
    }
  }
}

// ---------------- fused flash attention (no logits in HBM) ----------------
// grid = (HW/128, NH, B); 8 waves, each owns a 16-row Q tile; K/V async-staged 32 rows
// at a time with double buffering.
__global__ __launch_bounds__(256) void k_attn(
    const h16* __restrict__ Q, const h16* __restrict__ Kk, const h16* __restrict__ V,
    h16* __restrict__ O /* [B][HW][C] f16 */) {
  __shared__ h16 Ks[2][2048];
  __shared__ h16 Vs[2][2048];
  __shared__ h16 Ps[8 * 16 * 32];     // per-wave P tile bounce (C-layout -> A-layout)
  const int tid = threadIdx.x, lane = tid & 31, wave = tid >> 5;
  const int ml = lane & 15, hi = lane >> 4;
  const int h = blockIdx.y, b = blockIdx.z;
  const long long bh = (long long)b * NH + h;
  const h16* Qb = Q + bh * LP * HD;
  const h16* Kb = Kk + bh * LP * HD;
  const h16* Vb = V + bh * LP * HD;
  const int q0 = blockIdx.x * 128 + wave * 16;
  const int krow = (tid * 8) >> 6, dc = (tid * 8) & 63;

  auto stage = [&](int kt, int buf) {
    long long ga = (long long)(kt + krow) * HD + dc;
    async_ld_b128(Kb + ga, &Ks[buf][krow * 64 + dc]);
    async_ld_b128(Vb + ga, &Vs[buf][krow * 64 + dc]);
  };

  // Q fragments (16 rows x 64 dims -> two 16x32 A-frags), loaded once
  v16h a0, a1;
#pragma unroll
  for (int j = 0; j < 8; ++j) {
    int d = a_k0(j, hi);
    const h16* qp = Qb + (long long)(q0 + ml) * HD;
    a0[2 * j] = qp[d];      a0[2 * j + 1] = qp[d + 1];
    a1[2 * j] = qp[32 + d]; a1[2 * j + 1] = qp[32 + d + 1];
  }
  float mrun[8], lrun[8];
  v8f accO[4];
#pragma unroll
  for (int r = 0; r < 8; ++r) {
    mrun[r] = -1e30f; lrun[r] = 0.f;
#pragma unroll
    for (int g = 0; g < 4; ++g) accO[g][r] = 0.f;
  }

  const int nt = LP / 32;   // 35 key tiles
  stage(0, 0);
  wait_async0();
  __syncthreads();
  for (int t = 0; t < nt; ++t) {
    const int cur = t & 1, kt = t * 32;
    if (t + 1 < nt) stage(kt + 32, cur ^ 1);        // async prefetch next K/V tile
    const h16* Kt = Ks[cur];
    const h16* Vt = Vs[cur];

    // S = Q K^T over 32 keys -> two 16x16 C tiles
    v8f sarr[2];
#pragma unroll
    for (int t2 = 0; t2 < 2; ++t2) {
      v16h bk0, bk1;
#pragma unroll
      for (int j = 0; j < 8; ++j) {
        int kb = b_k0(j, hi);
        int base = (t2 * 16 + ml) * 64;             // key row in tile = N index
        bk0[2 * j] = Kt[base + kb];      bk0[2 * j + 1] = Kt[base + kb + 1];
        bk1[2 * j] = Kt[base + 32 + kb]; bk1[2 * j + 1] = Kt[base + 32 + kb + 1];
      }
      v8f c;
#pragma unroll
      for (int r = 0; r < 8; ++r) c[r] = 0.f;
      c = __builtin_amdgcn_wmma_f32_16x16x32_f16(false, a0, false, bk0, (short)0, c, false, false);
      c = __builtin_amdgcn_wmma_f32_16x16x32_f16(false, a1, false, bk1, (short)0, c, false, false);
      int kidx = kt + t2 * 16 + ml;
      bool valid = (kidx < LV);
#pragma unroll
      for (int r = 0; r < 8; ++r) sarr[t2][r] = valid ? c[r] * 0.125f : -1e30f;
    }
    // online softmax update (rows live in VGPR r of 16-lane groups)
#pragma unroll
    for (int r = 0; r < 8; ++r) {
      float v = fmaxf(sarr[0][r], sarr[1][r]);
#pragma unroll
      for (int msk = 1; msk < 16; msk <<= 1) v = fmaxf(v, __shfl_xor(v, msk, 32));
      float mnew = fmaxf(mrun[r], v);
      float scl = __expf(mrun[r] - mnew);
      mrun[r] = mnew;
      float p0 = __expf(sarr[0][r] - mnew);
      float p1 = __expf(sarr[1][r] - mnew);
      sarr[0][r] = p0; sarr[1][r] = p1;
      float sm = p0 + p1;
#pragma unroll
      for (int msk = 1; msk < 16; msk <<= 1) sm += __shfl_xor(sm, msk, 32);
      lrun[r] = lrun[r] * scl + sm;
#pragma unroll
      for (int g = 0; g < 4; ++g) accO[g][r] *= scl;
    }
    // bounce P (C-layout) through per-wave LDS into an A-fragment
    h16* Pw = Ps + wave * 16 * 32;
#pragma unroll
    for (int t2 = 0; t2 < 2; ++t2)
#pragma unroll
      for (int r = 0; r < 8; ++r)
        Pw[(r + 8 * hi) * 32 + t2 * 16 + ml] = (h16)sarr[t2][r];
    asm volatile("" ::: "memory");     // keep ds_store before ds_load (HW in-order per wave)
    v16h pa;
#pragma unroll
    for (int j = 0; j < 8; ++j) {
      int ka = a_k0(j, hi);
      pa[2 * j] = Pw[ml * 32 + ka]; pa[2 * j + 1] = Pw[ml * 32 + ka + 1];
    }
    // O += P V  (four 16-wide d-groups)
#pragma unroll
    for (int g = 0; g < 4; ++g) {
      v16h bv;
#pragma unroll
      for (int j = 0; j < 8; ++j) {
        int kb = b_k0(j, hi);
        bv[2 * j]     = Vt[kb * 64 + g * 16 + ml];
        bv[2 * j + 1] = Vt[(kb + 1) * 64 + g * 16 + ml];
      }
      accO[g] = __builtin_amdgcn_wmma_f32_16x16x32_f16(false, pa, false, bv, (short)0, accO[g], false, false);
    }
    wait_async0();
    __syncthreads();
  }
  // normalize + write O as f16 [B][HW][C] (head-concat layout for out-proj GEMM)
#pragma unroll
  for (int g = 0; g < 4; ++g)
#pragma unroll
    for (int r = 0; r < 8; ++r) {
      int row = q0 + r + 8 * hi;
      int col = h * HD + g * 16 + ml;
      float o = accO[g][r] / lrun[r];
      O[((long long)b * HWn + row) * CC + col] = (h16)o;
    }
}

// ---------------- host ----------------
extern "C" void kernel_launch(void* const* d_in, const int* in_sizes, int n_in,
                              void* d_out, int out_size, void* d_ws, size_t ws_size,
                              hipStream_t stream) {
  const float* x     = (const float*)d_in[0];
  const float* cond  = (const float*)d_in[1];
  const float* femb  = (const float*)d_in[2];
  const float* mod_w = (const float*)d_in[3];
  const float* mod_b = (const float*)d_in[4];
  const float* n1s   = (const float*)d_in[5];
  const float* n2s   = (const float*)d_in[6];
  const float* qkvsw = (const float*)d_in[7];
  const float* qkvcw = (const float*)d_in[8];
  const float* rfreq = (const float*)d_in[9];
  const float* qnw   = (const float*)d_in[10];
  const float* knw   = (const float*)d_in[11];
  const float* outw  = (const float*)d_in[12];
  const float* outb  = (const float*)d_in[13];
  const float* w1    = (const float*)d_in[14];
  const float* b1    = (const float*)d_in[15];
  const float* w2    = (const float*)d_in[16];
  const float* b2    = (const float*)d_in[17];
  float* out = (float*)d_out;

  char* wp = (char*)d_ws;
  auto alloc = [&](size_t bytes) -> void* {
    void* p = (void*)wp;
    wp += (bytes + 255) & ~(size_t)255;
    return p;
  };
  float* sfemb  = (float*)alloc((size_t)BB * EE * 4);
  float* modv   = (float*)alloc((size_t)BB * 6 * CC * 4);
  h16* hmod16   = (h16*)alloc((size_t)BB * CC * HWn * 2);
  h16* cond16   = (h16*)alloc((size_t)BB * SS * EE * 2);
  h16* wqs16    = (h16*)alloc((size_t)3 * CC * CC * 2);
  h16* wqc16    = (h16*)alloc((size_t)3 * CC * EE * 2);
  h16* wo16     = (h16*)alloc((size_t)CC * CC * 2);
  h16* w1_16    = (h16*)alloc((size_t)HID * CC * 2);
  h16* w2_16    = (h16*)alloc((size_t)CC * HID * 2);
  float* qkvs   = (float*)alloc((size_t)BB * HWn * 3 * CC * 4);   // [B][HW][3C]
  float* qkvc   = (float*)alloc((size_t)BB * SS * 3 * CC * 4);    // [B][S][3C]
  h16* Qh       = (h16*)alloc((size_t)BB * NH * LP * HD * 2);
  h16* Kh       = (h16*)alloc((size_t)BB * NH * LP * HD * 2);
  h16* Vh       = (h16*)alloc((size_t)BB * NH * LP * HD * 2);
  h16* Oh       = (h16*)alloc((size_t)BB * HWn * CC * 2);
  float* Y5     = (float*)alloc((size_t)BB * HWn * CC * 4);
  float* xn     = (float*)alloc((size_t)BB * CC * HWn * 4);
  h16* h2h      = (h16*)alloc((size_t)BB * CC * HWn * 2);
  h16* g6h      = (h16*)alloc((size_t)BB * HWn * HID * 2);
  float* m2     = (float*)alloc((size_t)BB * HWn * CC * 4);

  auto cvt = [&](const float* src, h16* dst, int n) {
    k_cvt_f32_f16<<<(n + 255) / 256, 256, 0, stream>>>(src, dst, n);
  };
  cvt(qkvsw, wqs16, 3 * CC * CC);
  cvt(qkvcw, wqc16, 3 * CC * EE);
  cvt(outw,  wo16,  CC * CC);
  cvt(w1,    w1_16, HID * CC);
  cvt(w2,    w2_16, CC * HID);
  cvt(cond,  cond16, BB * SS * EE);

  // adaLN modulation
  k_silu<<<(BB * EE + 255) / 256, 256, 0, stream>>>(femb, sfemb, BB * EE);
  k_mod<<<(BB * 6 * CC + 255) / 256, 256, 0, stream>>>(sfemb, mod_w, mod_b, modv);

  // norm1 + modulate -> h_mod f16 [B][C][HW]
  k_normmod<<<(BB * HWn + 255) / 256, 256, 0, stream>>>(x, n1s, modv, 0, CC, hmod16);

  // qkv_s[b] = hmod[b]^T[1024,512] x W_s^T[512->1536]  (A KM, B NK) -> [B][HW][3C]
  gemm_wmma<1, 1, 0><<<dim3(3 * CC / 64, HWn / 64, BB), 256, 0, stream>>>(
      hmod16, (long long)CC * HWn, wqs16, 0ll, qkvs, (long long)HWn * 3 * CC,
      (h16*)nullptr, 0ll, (const float*)nullptr, HWn, 3 * CC, CC);

  // qkv_c[b] = cond[b][77,768] x Wc^T[768->1536]       (A MK, B NK) -> [B][S][3C]
  gemm_wmma<0, 1, 0><<<dim3(3 * CC / 64, (SS + 63) / 64, BB), 256, 0, stream>>>(
      cond16, (long long)SS * EE, wqc16, 0ll, qkvc, (long long)SS * 3 * CC,
      (h16*)nullptr, 0ll, (const float*)nullptr, SS, 3 * CC, EE);

  // RoPE + QK-RMSNorm -> padded f16 Q/K/V
  {
    int waves = BB * NH * LP;
    k_build_qkv<<<(waves * 32 + 255) / 256, 256, 0, stream>>>(
        qkvs, qkvc, rfreq, qnw, knw, Qh, Kh, Vh);
  }

  // fused flash attention -> Oh f16 [B][HW][C]
  k_attn<<<dim3(HWn / 128, NH, BB), 256, 0, stream>>>(Qh, Kh, Vh, Oh);

  // out-proj: Y5[b] = Oh[b][1024,512] x out_w^T + out_b   (A MK, B NK, bias)
  gemm_wmma<0, 1, 1><<<dim3(CC / 64, HWn / 64, BB), 256, 0, stream>>>(
      Oh, (long long)HWn * CC, wo16, 0ll, Y5, (long long)HWn * CC,
      (h16*)nullptr, 0ll, outb, HWn, CC, CC);

  // x = x + g1 * attn_out
  k_resid<<<(int)(((long long)BB * CC * HWn + 255) / 256), 256, 0, stream>>>(
      x, modv, 2 * CC, Y5, xn);

  // norm2 + modulate -> h2 f16 [B][C][HW]
  k_normmod<<<(BB * HWn + 255) / 256, 256, 0, stream>>>(xn, n2s, modv, 3 * CC, 4 * CC, h2h);

  // MLP1: g6h[b] = gelu(h2[b]^T[1024,512] x W1^T + b1) f16   (A KM, B NK, bias+gelu)
  gemm_wmma<1, 1, 2><<<dim3(HID / 64, HWn / 64, BB), 256, 0, stream>>>(
      h2h, (long long)CC * HWn, w1_16, 0ll, (float*)nullptr, 0ll,
      g6h, (long long)HWn * HID, b1, HWn, HID, CC);

  // MLP2: m2[b] = g6h[b][1024,2048] x W2^T + b2   (A MK, B NK, bias)
  gemm_wmma<0, 1, 1><<<dim3(CC / 64, HWn / 64, BB), 256, 0, stream>>>(
      g6h, (long long)HWn * HID, w2_16, 0ll, m2, (long long)HWn * CC,
      (h16*)nullptr, 0ll, b2, HWn, CC, HID);

  // out = x + g2 * mlp_out
  k_resid<<<(int)(((long long)BB * CC * HWn + 255) / 256), 256, 0, stream>>>(
      xn, modv, 5 * CC, m2, out);

  (void)in_sizes; (void)n_in; (void)out_size; (void)ws_size;
}
